// LinearAttention_21517786153021
// MI455X (gfx1250) — matrix operand order repaired and checked by reading, not compiled, not run on hardware
//
#include <hip/hip_runtime.h>
#include <hip/hip_bf16.h>
#include <math.h>

// ---------------------------------------------------------------------------
// Linear attention, single query per batch, on gfx1250 (MI455X).
//   out[b,v] = (sum_l s_l * V[b,l,v]) / (sum_l s_l + eps)
//   s_l      = phi(emb[k[b,l]] @ Wk + bk) . phi(emb[q[b]] @ Wq + bq)
// The 512x256 KV state is algebraically eliminated; the only large GEMM is
// the K projection, done with v_wmma_f32_16x16x32_bf16 (f32 accumulate).
// ---------------------------------------------------------------------------

typedef __bf16 bf16_t;
typedef __attribute__((ext_vector_type(16))) __bf16 v16bf;
typedef __attribute__((ext_vector_type(8)))  float  v8f;

#define B_      32
#define L_      4096
#define EMBED_  512
#define DKEY_   256
#define DV_     512
#define ROWS_PER_WG 128   // 8 waves x 16 rows
#define CHUNKS  (L_ / ROWS_PER_WG)   // 32 chunks per batch

// Workspace layout (bytes)
#define WS_NUM_OFF   0                     // 32*512 f32   = 65536
#define WS_DEN_OFF   65536                 // 32 f32       = 128
#define WS_QV_OFF    65792                 // 32*256 f32   = 32768
#define WS_WKT_OFF   98560                 // 512*256 bf16 = 262144 (Wk^T, [n][k])

// ---------------------------------------------------------------------------
// Convert Wk [EMBED,DKEY] f32 (k-major) -> Wk^T [DKEY][EMBED] bf16 (n-major)
// so B-fragments are contiguous in K.
// ---------------------------------------------------------------------------
__global__ void convert_wk_kernel(const float* __restrict__ Wk,
                                  bf16_t* __restrict__ WkT) {
    const int e  = blockIdx.x * 256 + threadIdx.x;   // 131072 elements
    const int kk = e >> 8;      // 0..511
    const int n  = e & 255;     // 0..255
    WkT[(size_t)n * EMBED_ + kk] = (bf16_t)Wk[e];
}

// ---------------------------------------------------------------------------
// Q projection: Qv[b,n] = phi(emb[q[b]] . Wq[:,n] + bq[n]).  Tiny (8.4 MFLOP),
// full f32 for accuracy. 32 blocks x 256 threads.
// ---------------------------------------------------------------------------
__global__ void qproj_kernel(const int*   __restrict__ q,
                             const float* __restrict__ emb,
                             const float* __restrict__ Wq,
                             const float* __restrict__ bq,
                             float* __restrict__ Qv) {
    const int b = blockIdx.x;
    const int n = threadIdx.x;
    const float* __restrict__ qe = emb + (size_t)q[b] * EMBED_;
    float s = bq[n];
    for (int kk = 0; kk < EMBED_; ++kk)
        s = fmaf(qe[kk], Wq[kk * DKEY_ + n], s);
    Qv[b * DKEY_ + n] = (s > 0.f) ? (s + 1.f) : __expf(s);  // elu(x)+1
}

// ---------------------------------------------------------------------------
// Main fused kernel. One WG = one (batch, 128-row L chunk). 8 waves, each wave
// projects 16 gathered rows to 256 key dims via 16 WMMA accumulator tiles.
// ---------------------------------------------------------------------------
__global__ __launch_bounds__(256)
void lin_attn_main_kernel(const int*   __restrict__ kidx,
                          const float* __restrict__ v_emb,
                          const float* __restrict__ emb_table,
                          const bf16_t* __restrict__ WkT,
                          const float* __restrict__ Wk_b,
                          const float* __restrict__ Qv,
                          float* __restrict__ ws_num,
                          float* __restrict__ ws_den) {
    __shared__ bf16_t Bsh[DKEY_ * 32];     // 16 KB: Wk^T slice [256 n][32 k]
    __shared__ float  num_sh[DV_];         // per-WG numerator partial
    __shared__ float  s_sh[8 * 16];        // per-row scores
    __shared__ float  den_sh;

    const int wg    = blockIdx.x;
    const int b     = wg >> 5;             // 32 chunks per batch
    const int chunk = wg & 31;
    const int l0    = chunk * ROWS_PER_WG;
    const int tid   = threadIdx.x;
    const int wave  = tid >> 5;
    const int lane  = tid & 31;
    const int half  = lane >> 4;           // 0: lanes 0-15, 1: lanes 16-31
    const int l16   = lane & 15;

    for (int i = tid; i < DV_; i += 256) num_sh[i] = 0.f;
    if (tid == 0) den_sh = 0.f;

    // Each lane owns one gathered A row (row = wave*16 + l16; both halves
    // of the wave cover the same 16 rows per the ISA 16-bit A layout).
    const int row_in_wg = wave * 16 + l16;
    const int kid = kidx[b * L_ + l0 + row_in_wg];
    const float* __restrict__ arow = emb_table + (size_t)kid * EMBED_;

    v8f acc[16] = {};   // 16 N-tiles of 16x16 f32 accumulators (128 VGPRs)

    for (int ks = 0; ks < 16; ++ks) {      // K = 512 in steps of 32
        __syncthreads();                   // prior Bsh readers done
        // Stage Wk^T[:, ks*32 .. +31] : thread t stages column-tile row n=t.
        {
            const uint4* __restrict__ src =
                (const uint4*)(WkT + (size_t)tid * EMBED_ + ks * 32);
            uint4* dst = (uint4*)(Bsh + tid * 32);
            dst[0] = src[0]; dst[1] = src[1]; dst[2] = src[2]; dst[3] = src[3];
        }
        __syncthreads();

        // --- A fragment: ISA layout for 16-bit 16x32 A ---
        // lanes 0-15 : K = {k0..k0+7, k0+16..k0+23}
        // lanes 16-31: K = {k0+8..k0+15, k0+24..k0+31}
        const int kbase = ks * 32 + half * 8;
        const float4 a0 = *(const float4*)(arow + kbase);
        const float4 a1 = *(const float4*)(arow + kbase + 4);
        const float4 a2 = *(const float4*)(arow + kbase + 16);
        const float4 a3 = *(const float4*)(arow + kbase + 20);
        if (ks < 15) __builtin_prefetch(arow + (ks + 1) * 32 + half * 8, 0, 1);

        v16bf afrag;
        afrag[0]=(bf16_t)a0.x; afrag[1]=(bf16_t)a0.y; afrag[2]=(bf16_t)a0.z; afrag[3]=(bf16_t)a0.w;
        afrag[4]=(bf16_t)a1.x; afrag[5]=(bf16_t)a1.y; afrag[6]=(bf16_t)a1.z; afrag[7]=(bf16_t)a1.w;
        afrag[8]=(bf16_t)a2.x; afrag[9]=(bf16_t)a2.y; afrag[10]=(bf16_t)a2.z; afrag[11]=(bf16_t)a2.w;
        afrag[12]=(bf16_t)a3.x; afrag[13]=(bf16_t)a3.y; afrag[14]=(bf16_t)a3.z; afrag[15]=(bf16_t)a3.w;

        // --- 16 N-tiles: B fragment from LDS, then WMMA ---
        // B layout (32x16, 16-bit): lane n<16 holds col n, K = {0..7, 16..23};
        // lanes 16-31 hold K = {8..15, 24..31}.
        #pragma unroll
        for (int nt = 0; nt < 16; ++nt) {
            const bf16_t* __restrict__ bp = Bsh + (nt * 16 + l16) * 32 + half * 8;
            v16bf bfrag;
            #pragma unroll
            for (int i = 0; i < 16; ++i) bfrag[i] = bp[(i < 8) ? i : i + 8];
            acc[nt] = __builtin_amdgcn_wmma_f32_16x16x32_bf16(
                false, afrag, false, bfrag, (short)0, acc[nt], false, false);
        }
    }

    // --- Epilogue: bias + phi, dot with Q -> per-row score partials ---
    // C/D layout: acc[nt][j] = row (half?j+8:j), col nt*16+l16.
    float p[8] = {0.f,0.f,0.f,0.f,0.f,0.f,0.f,0.f};
    #pragma unroll
    for (int nt = 0; nt < 16; ++nt) {
        const int col = nt * 16 + l16;
        const float bias = Wk_b[col];
        const float qc   = Qv[b * DKEY_ + col];
        #pragma unroll
        for (int j = 0; j < 8; ++j) {
            const float x  = acc[nt][j] + bias;
            const float kv = (x > 0.f) ? (x + 1.f) : __expf(x);  // elu+1
            p[j] = fmaf(kv, qc, p[j]);
        }
    }
    // reduce across the 16 lanes of each half-wave (cols)
    #pragma unroll
    for (int j = 0; j < 8; ++j) {
        p[j] += __shfl_xor(p[j], 1, 32);
        p[j] += __shfl_xor(p[j], 2, 32);
        p[j] += __shfl_xor(p[j], 4, 32);
        p[j] += __shfl_xor(p[j], 8, 32);
    }
    if (l16 == 0) {   // lane 0 -> rows 0..7, lane 16 -> rows 8..15
        float hd = 0.f;
        #pragma unroll
        for (int j = 0; j < 8; ++j) { s_sh[wave * 16 + half * 8 + j] = p[j]; hd += p[j]; }
        atomicAdd(&den_sh, hd);
    }
    __syncthreads();

    // --- num += sum_r s_r * V[row_r, :]   (V read exactly once) ---
    const float* __restrict__ Vbase =
        v_emb + ((size_t)b * L_ + l0 + wave * 16) * DV_;
    float sv[16];
    #pragma unroll
    for (int r = 0; r < 16; ++r) sv[r] = s_sh[wave * 16 + r];

    #pragma unroll 4
    for (int i = 0; i < 16; ++i) {
        const int v = lane + i * 32;
        float a = 0.f;
        #pragma unroll
        for (int r = 0; r < 16; ++r)
            a = fmaf(sv[r], Vbase[(size_t)r * DV_ + v], a);
        atomicAdd(&num_sh[v], a);   // ds_add_f32
    }
    __syncthreads();

    // --- flush WG partials to global accumulators ---
    atomicAdd(&ws_num[b * DV_ + tid],       num_sh[tid]);
    atomicAdd(&ws_num[b * DV_ + tid + 256], num_sh[tid + 256]);
    if (tid == 0) atomicAdd(&ws_den[b], den_sh);
}

// ---------------------------------------------------------------------------
// out[b,v] = num[b,v] / (den[b] + eps)
// ---------------------------------------------------------------------------
__global__ void finalize_kernel(const float* __restrict__ num,
                                const float* __restrict__ den,
                                float* __restrict__ out) {
    const int b = blockIdx.x;
    const int v = threadIdx.x;
    out[b * DV_ + v] = num[b * DV_ + v] / (den[b] + 1e-6f);
}

extern "C" void kernel_launch(void* const* d_in, const int* in_sizes, int n_in,
                              void* d_out, int out_size, void* d_ws, size_t ws_size,
                              hipStream_t stream) {
    const int*   k     = (const int*)  d_in[0];
    const int*   q     = (const int*)  d_in[1];
    const float* v_emb = (const float*)d_in[2];
    const float* emb   = (const float*)d_in[3];
    const float* Wk_w  = (const float*)d_in[4];
    const float* Wk_b  = (const float*)d_in[5];
    const float* Wq_w  = (const float*)d_in[6];
    const float* Wq_b  = (const float*)d_in[7];
    float* out = (float*)d_out;

    char* ws = (char*)d_ws;
    float*  ws_num = (float*)(ws + WS_NUM_OFF);
    float*  ws_den = (float*)(ws + WS_DEN_OFF);
    float*  Qv     = (float*)(ws + WS_QV_OFF);
    bf16_t* WkT    = (bf16_t*)(ws + WS_WKT_OFF);

    // zero the global accumulators (num + den) every call
    hipMemsetAsync(ws, 0, WS_DEN_OFF + 256, stream);

    convert_wk_kernel<<<512, 256, 0, stream>>>(Wk_w, WkT);
    qproj_kernel<<<B_, DKEY_, 0, stream>>>(q, emb, Wq_w, Wq_b, Qv);
    lin_attn_main_kernel<<<B_ * CHUNKS, 256, 0, stream>>>(
        k, v_emb, emb, WkT, Wk_b, Qv, ws_num, ws_den);
    finalize_kernel<<<B_, DV_, 0, stream>>>(ws_num, ws_den, out);
}
